// TransformerBlock_9543417332375
// MI455X (gfx1250) — matrix operand order repaired and checked by reading, not compiled, not run on hardware
//
#include <hip/hip_runtime.h>
#include <math.h>

// ---------------------------------------------------------------------------
// CDNA5 (gfx1250) transformer block: LN -> QKV GEMM -> masked attention
// (+ mean-probs output) -> out-proj(+residual,padmask) -> LN -> FFN(quickGELU)
// All matmuls via v_wmma_f32_16x16x32_bf16 (wave32). Global->LDS staging uses
// gfx1250 async-to-LDS when the toolchain exposes the builtin.
// ---------------------------------------------------------------------------

typedef __attribute__((ext_vector_type(16))) __bf16 v16bf;
typedef __attribute__((ext_vector_type(8)))  float  v8f;
typedef __attribute__((vector_size(4 * sizeof(int)))) int v4i_t;

union FragBF {
  unsigned int u[8];
  v16bf v;
};

// ---- optional gfx1250 async global->LDS path ------------------------------
#if defined(__has_builtin)
#if __has_builtin(__builtin_amdgcn_global_load_async_to_lds_b128) && \
    __has_builtin(__builtin_amdgcn_s_wait_asynccnt)
#define USE_ASYNC_LDS 1
#endif
#endif
#ifndef USE_ASYNC_LDS
#define USE_ASYNC_LDS 0
#endif

#if USE_ASYNC_LDS
__device__ __forceinline__ void copy_b128_to_lds(const void* g, void* l) {
  // prototype (from hipcc diagnostic): param is `int __vector(4) *`
  __builtin_amdgcn_global_load_async_to_lds_b128(
      (v4i_t*)g, (v4i_t*)l, /*offset=*/0, /*cpol=*/0);
}
__device__ __forceinline__ void wait_async_all() {
  __builtin_amdgcn_s_wait_asynccnt(0);
}
#else
__device__ __forceinline__ void copy_b128_to_lds(const void* g, void* l) {
  *(uint4*)l = *(const uint4*)g;
}
__device__ __forceinline__ void wait_async_all() {}
#endif

__device__ __forceinline__ unsigned short f32_to_bf16(float f) {
  unsigned int u = __builtin_bit_cast(unsigned int, f);
  unsigned int r = u + 0x7FFFu + ((u >> 16) & 1u);   // round-to-nearest-even
  return (unsigned short)(r >> 16);
}

__device__ __forceinline__ v8f wmma_bf16(v16bf a, v16bf b, v8f c) {
  return __builtin_amdgcn_wmma_f32_16x16x32_bf16(false, a, false, b,
                                                 (short)0, c, false, false);
}

// A-fragment (16x32 bf16): lane L: m=L&15, g=L>>4. VGPR v holds
// K = (v<4 ? 2v : 16+2(v-4)) + 8g + {0,1}. Contiguous pairs -> ds_load_b128.
__device__ __forceinline__ v16bf load_fragA(const unsigned short* base,
                                            int stride, int lane) {
  const int g = (lane >> 4) & 1, m = lane & 15;
  FragBF f;
#pragma unroll
  for (int vv = 0; vv < 8; ++vv) {
    const int k = (vv < 4 ? 2 * vv : 16 + 2 * (vv - 4)) + 8 * g;
    f.u[vv] = *(const unsigned int*)(base + m * stride + k);
  }
  return f.v;
}

// B-fragment (32x16 bf16), B[k][n] = base[n*stride + k]; lane L: n=L&15,
// g=L>>4. VGPR v holds K = (v<4 ? 2v : 16+2(v-4)) + 8g + {0,1}.
__device__ __forceinline__ v16bf load_fragB(const unsigned short* base,
                                            int stride, int lane) {
  const int g = (lane >> 4) & 1, n = lane & 15;
  FragBF f;
#pragma unroll
  for (int vv = 0; vv < 8; ++vv) {
    const int k = (vv < 4 ? 2 * vv : 16 + 2 * (vv - 4)) + 8 * g;
    f.u[vv] = *(const unsigned int*)(base + n * stride + k);
  }
  return f.v;
}

// ---------------------------------------------------------------------------
// float -> bf16 conversion (weights)
// ---------------------------------------------------------------------------
__global__ void __launch_bounds__(256) cvt_f32_bf16(const float* __restrict__ in,
                                                    unsigned short* __restrict__ out,
                                                    size_t n) {
  size_t i = (size_t)blockIdx.x * blockDim.x + threadIdx.x;
  const size_t stride = (size_t)gridDim.x * blockDim.x;
  for (; i < n; i += stride) out[i] = f32_to_bf16(in[i]);
}

// ---------------------------------------------------------------------------
// LayerNorm (fp32 stats) -> bf16 output. One 256-thread block per row.
// ---------------------------------------------------------------------------
__global__ void __launch_bounds__(256) layernorm_bf16(const float* __restrict__ x,
                                                      const float* __restrict__ gam,
                                                      const float* __restrict__ bet,
                                                      unsigned short* __restrict__ out,
                                                      int Dd) {
  __shared__ float s1[256], s2[256];
  const int row = blockIdx.x;
  const int t = threadIdx.x;
  const float* xr = x + (size_t)row * Dd;
  float a = 0.f, b = 0.f;
  for (int i = t; i < Dd; i += 256) {
    const float v = xr[i];
    a += v;
    b += v * v;
  }
  s1[t] = a; s2[t] = b;
  __syncthreads();
  for (int s = 128; s > 0; s >>= 1) {
    if (t < s) { s1[t] += s1[t + s]; s2[t] += s2[t + s]; }
    __syncthreads();
  }
  const float mean = s1[0] / Dd;
  const float var = s2[0] / Dd - mean * mean;
  const float inv = rsqrtf(var + 1e-5f);
  for (int i = t; i < Dd; i += 256) {
    const float v = (xr[i] - mean) * inv * gam[i] + bet[i];
    out[(size_t)row * Dd + i] = f32_to_bf16(v);
  }
}

// ---------------------------------------------------------------------------
// Tiled bf16 GEMM:  C[M,Nc] = A[M,K] @ W[Nc,K]^T + bias, fused epilogues.
//   mode 0: store bf16
//   mode 1: quickGELU, store bf16
//   mode 2: fp32 out = resid + (qpad[m] ? 0 : C)    (out-proj)
//   mode 3: fp32 out = resid + C                     (FFN2)
// Block: 256 threads (8 waves), 64(M)x128(N) tile, K-step 32.
// Double-buffered LDS, one barrier per K-step, async global->LDS staging.
// Each wave: 1 A-frag reused over 4 B-frags -> 4 WMMA per K-step.
// ---------------------------------------------------------------------------
__global__ void __launch_bounds__(256) gemm_bf16_kernel(
    const unsigned short* __restrict__ A,
    const unsigned short* __restrict__ W,
    const float* __restrict__ bias,
    int M, int Nc, int K, int mode,
    unsigned short* __restrict__ outb,
    float* __restrict__ outf,
    const float* __restrict__ resid,
    const unsigned char* __restrict__ qpad) {
  __shared__ unsigned short As[2][64][40];    // 80B row stride (16B aligned)
  __shared__ unsigned short Ws[2][128][40];

  const int t = threadIdx.x;
  const int lane = t & 31;
  const int wv = t >> 5;
  const int g = (lane >> 4) & 1, ln = lane & 15;
  const int msub = wv & 3, nh = wv >> 2;       // 4 M-strips x 2 N-halves
  const int m0 = blockIdx.y * 64, n0 = blockIdx.x * 128;
  const int arow = t >> 2, acol = (t & 3) * 8;   // A: 1x b128 / thread
  const int wrow = t >> 1, wcol = (t & 1) * 16;  // W: 2x b128 / thread

  auto stage_tiles = [&](int k0, int s) {
    const unsigned short* ga = A + (size_t)(m0 + arow) * K + k0 + acol;
    const unsigned short* gw = W + (size_t)(n0 + wrow) * K + k0 + wcol;
    copy_b128_to_lds(ga, &As[s][arow][acol]);
    copy_b128_to_lds(gw, &Ws[s][wrow][wcol]);
    copy_b128_to_lds(gw + 8, &Ws[s][wrow][wcol + 8]);
    if (k0 + 32 < K) {  // keep next K-tile warm in near caches
      __builtin_prefetch(ga + 32, 0, 3);
      __builtin_prefetch(gw + 32, 0, 3);
    }
  };

  v8f acc[4] = {};
  const int nk = K / 32;
  stage_tiles(0, 0);
  for (int kt = 0; kt < nk; ++kt) {
    const int cur = kt & 1;
    wait_async_all();        // publish stage `cur`
    __syncthreads();
    if (kt + 1 < nk) stage_tiles((kt + 1) * 32, cur ^ 1);
    const v16bf a = load_fragA(&As[cur][msub * 16][0], 40, lane);
#pragma unroll
    for (int j = 0; j < 4; ++j) {
      const v16bf bb = load_fragB(&Ws[cur][nh * 64 + j * 16][0], 40, lane);
      acc[j] = wmma_bf16(a, bb, acc[j]);
    }
  }

#pragma unroll
  for (int r = 0; r < 8; ++r) {
    const int mloc = msub * 16 + r + 8 * g;
    const size_t m = (size_t)m0 + mloc;
#pragma unroll
    for (int j = 0; j < 4; ++j) {
      const int n = n0 + nh * 64 + j * 16 + ln;
      float v = acc[j][r] + bias[n];
      if (mode == 0) {
        outb[m * Nc + n] = f32_to_bf16(v);
      } else if (mode == 1) {
        v = v / (1.f + expf(-1.702f * v));   // quickGELU
        outb[m * Nc + n] = f32_to_bf16(v);
      } else if (mode == 2) {
        const float p = qpad[m] ? 0.f : 1.f;
        outf[m * Nc + n] = resid[m * Nc + n] + p * v;
      } else {
        outf[m * Nc + n] = resid[m * Nc + n] + v;
      }
    }
  }
}

// ---------------------------------------------------------------------------
// Attention: one block per (b, h, 64-query tile). Causal + key-pad mask with
// diagonal always allowed. Three passes over k-tiles:
//   1) row max  2) row sum(exp)  3) normalized P -> weights (atomic, /H),
//      P (bf16 in LDS) @ V via WMMA.
// ---------------------------------------------------------------------------
#define HDIM 64

__global__ void __launch_bounds__(256) attention_kernel(
    const unsigned short* __restrict__ qkv,   // [B*N, 3D] bf16
    const unsigned char* __restrict__ kpm,    // [B*N] key padding
    unsigned short* __restrict__ attn,        // [B*N, D] bf16
    float* __restrict__ weights,              // [B, N, N] fp32 (pre-zeroed)
    int Nn, int Dd, int Hh) {
  __shared__ unsigned short Qs[64][72];
  __shared__ unsigned short Ks[64][72];
  __shared__ unsigned short Vt[64][72];       // transposed: Vt[d][k]
  __shared__ unsigned short Ps[64][72];
  __shared__ float mrow[64], lrow[64];
  __shared__ float red[2][64];
  __shared__ unsigned char padk[64];

  const int qt = blockIdx.x, h = blockIdx.y, b = blockIdx.z;
  const int q0 = qt * 64;
  const int t = threadIdx.x, lane = t & 31, wv = t >> 5;
  const int g = (lane >> 4) & 1, ln = lane & 15;
  const int msub = wv & 3, kp = wv >> 2;
  const int row3 = t >> 2, c16 = (t & 3) * 16;
  const size_t base = (size_t)b * Nn;
  const int W3 = 3 * Dd;

  // Q tile (stays resident)
  {
    const unsigned short* src = qkv + (base + q0 + row3) * W3 + h * HDIM + c16;
    copy_b128_to_lds(src, &Qs[row3][c16]);
    copy_b128_to_lds(src + 8, &Qs[row3][c16 + 8]);
  }

  auto load_k = [&](int kt) {
    const unsigned short* src =
        qkv + (base + (size_t)kt * 64 + row3) * W3 + Dd + h * HDIM + c16;
    copy_b128_to_lds(src, &Ks[row3][c16]);
    copy_b128_to_lds(src + 8, &Ks[row3][c16 + 8]);
    if (t < 64) padk[t] = kpm[base + (size_t)kt * 64 + t];
  };
  auto load_v = [&](int kt) {  // transpose on store -> contiguous K pairs
    const unsigned short* src =
        qkv + (base + (size_t)kt * 64 + row3) * W3 + 2 * Dd + h * HDIM + c16;
    uint4 d0 = *(const uint4*)(src);
    uint4 d1 = *(const uint4*)(src + 8);
    const unsigned short* e0 = (const unsigned short*)&d0;
    const unsigned short* e1 = (const unsigned short*)&d1;
#pragma unroll
    for (int j = 0; j < 8; ++j) Vt[c16 + j][row3] = e0[j];
#pragma unroll
    for (int j = 0; j < 8; ++j) Vt[c16 + 8 + j][row3] = e1[j];
  };

  // masked, scaled scores for one 16x16 subtile (ksub in {2kp, 2kp+1})
  auto scores = [&](int kt, int ksub, float* sv) {
    v8f s = {};
#pragma unroll
    for (int st = 0; st < 2; ++st) {
      const v16bf a  = load_fragA(&Qs[msub * 16][32 * st], 72, lane);
      const v16bf bb = load_fragB(&Ks[ksub * 16][32 * st], 72, lane);
      s = wmma_bf16(a, bb, s);
    }
    const int kloc = ksub * 16 + ln;
    const int kg = kt * 64 + kloc;
    const bool kpad = padk[kloc] != 0;
#pragma unroll
    for (int r = 0; r < 8; ++r) {
      const int qg = q0 + msub * 16 + r + 8 * g;
      const bool allowed = (kg == qg) || ((kg < qg) && !kpad);
      sv[r] = allowed ? s[r] * 0.125f : -__builtin_inff();
    }
  };

  // ---- pass 1: row max ----------------------------------------------------
  float rmax[8];
#pragma unroll
  for (int r = 0; r < 8; ++r) rmax[r] = -__builtin_inff();
  for (int kt = 0; kt <= qt; ++kt) {
    load_k(kt);
    wait_async_all();
    __syncthreads();
    float sv0[8], sv1[8];
    scores(kt, kp * 2, sv0);
    scores(kt, kp * 2 + 1, sv1);
#pragma unroll
    for (int r = 0; r < 8; ++r)
      rmax[r] = fmaxf(rmax[r], fmaxf(sv0[r], sv1[r]));
    __syncthreads();
  }
#pragma unroll
  for (int r = 0; r < 8; ++r) {
    float v = rmax[r];
    v = fmaxf(v, __shfl_xor(v, 1));
    v = fmaxf(v, __shfl_xor(v, 2));
    v = fmaxf(v, __shfl_xor(v, 4));
    v = fmaxf(v, __shfl_xor(v, 8));
    if (ln == 0) red[kp][msub * 16 + r + 8 * g] = v;
  }
  __syncthreads();
  if (t < 64) mrow[t] = fmaxf(red[0][t], red[1][t]);
  __syncthreads();

  // ---- pass 2: row sum ----------------------------------------------------
  float rsum[8];
#pragma unroll
  for (int r = 0; r < 8; ++r) rsum[r] = 0.f;
  for (int kt = 0; kt <= qt; ++kt) {
    load_k(kt);
    wait_async_all();
    __syncthreads();
    float sv0[8], sv1[8];
    scores(kt, kp * 2, sv0);
    scores(kt, kp * 2 + 1, sv1);
#pragma unroll
    for (int r = 0; r < 8; ++r) {
      const float mm = mrow[msub * 16 + r + 8 * g];
      rsum[r] += expf(sv0[r] - mm) + expf(sv1[r] - mm);
    }
    __syncthreads();
  }
#pragma unroll
  for (int r = 0; r < 8; ++r) {
    float v = rsum[r];
    v += __shfl_xor(v, 1);
    v += __shfl_xor(v, 2);
    v += __shfl_xor(v, 4);
    v += __shfl_xor(v, 8);
    if (ln == 0) red[kp][msub * 16 + r + 8 * g] = v;
  }
  __syncthreads();
  if (t < 64) lrow[t] = red[0][t] + red[1][t];
  __syncthreads();

  // ---- pass 3: P -> weights, O += P @ V -----------------------------------
  v8f o0 = {}, o1 = {};
  const float invH = 1.0f / Hh;
  for (int kt = 0; kt <= qt; ++kt) {
    load_k(kt);
    load_v(kt);
    wait_async_all();
    __syncthreads();
    float sv0[8], sv1[8];
    scores(kt, kp * 2, sv0);
    scores(kt, kp * 2 + 1, sv1);
#pragma unroll
    for (int r = 0; r < 8; ++r) {
      const int mloc = msub * 16 + r + 8 * g;
      const float mm = mrow[mloc];
      const float il = 1.0f / lrow[mloc];
      const float p0 = expf(sv0[r] - mm) * il;
      const float p1 = expf(sv1[r] - mm) * il;
      const int qg = q0 + mloc;
      const int kg0 = kt * 64 + (kp * 2) * 16 + ln;
      const size_t widx = ((size_t)b * Nn + qg) * Nn;
      atomicAdd(&weights[widx + kg0], p0 * invH);
      atomicAdd(&weights[widx + kg0 + 16], p1 * invH);
      Ps[mloc][(kp * 2) * 16 + ln]     = f32_to_bf16(p0);
      Ps[mloc][(kp * 2 + 1) * 16 + ln] = f32_to_bf16(p1);
    }
    __syncthreads();
#pragma unroll
    for (int st = 0; st < 2; ++st) {
      const v16bf a  = load_fragA(&Ps[msub * 16][32 * st], 72, lane);
      const v16bf b0 = load_fragB(&Vt[(kp * 2 + 0) * 16][32 * st], 72, lane);
      const v16bf b1 = load_fragB(&Vt[(kp * 2 + 1) * 16][32 * st], 72, lane);
      o0 = wmma_bf16(a, b0, o0);
      o1 = wmma_bf16(a, b1, o1);
    }
    __syncthreads();
  }

#pragma unroll
  for (int r = 0; r < 8; ++r) {
    const int mloc = msub * 16 + r + 8 * g;
    const size_t orow = (base + q0 + mloc) * Dd + h * HDIM;
    attn[orow + (kp * 2) * 16 + ln]      = f32_to_bf16(o0[r]);
    attn[orow + (kp * 2 + 1) * 16 + ln]  = f32_to_bf16(o1[r]);
  }
}

// ---------------------------------------------------------------------------
// Host orchestration
// ---------------------------------------------------------------------------
extern "C" void kernel_launch(void* const* d_in, const int* in_sizes, int n_in,
                              void* d_out, int out_size, void* d_ws,
                              size_t ws_size, hipStream_t stream) {
  (void)in_sizes; (void)n_in; (void)out_size; (void)ws_size;
  const int Bb = 2, Nn = 2048, Dd = 1024, Hh = 16, Ff = 4096;
  const int M = Bb * Nn;  // 4096 tokens

  const float* x            = (const float*)d_in[0];
  const unsigned char* kpm  = (const unsigned char*)d_in[1];  // bool mask
  const float* ln1g = (const float*)d_in[2];
  const float* ln1b = (const float*)d_in[3];
  const float* w_in = (const float*)d_in[4];
  const float* b_in = (const float*)d_in[5];
  const float* w_out = (const float*)d_in[6];
  const float* b_out = (const float*)d_in[7];
  const float* ln2g = (const float*)d_in[8];
  const float* ln2b = (const float*)d_in[9];
  const float* w1 = (const float*)d_in[10];
  const float* b1 = (const float*)d_in[11];
  const float* w2 = (const float*)d_in[12];
  const float* b2 = (const float*)d_in[13];

  float* out_x = (float*)d_out;                       // [B,N,D]
  float* out_w = out_x + (size_t)M * Dd;              // [B,N,N]

  // workspace carve-out (256B aligned)
  char* ws = (char*)d_ws;
  size_t off = 0;
  auto alloc = [&](size_t bytes) -> void* {
    void* p = ws + off;
    off += (bytes + 255) & ~(size_t)255;
    return p;
  };
  unsigned short* w_in_bf  = (unsigned short*)alloc((size_t)3 * Dd * Dd * 2);
  unsigned short* w_out_bf = (unsigned short*)alloc((size_t)Dd * Dd * 2);
  unsigned short* w1_bf    = (unsigned short*)alloc((size_t)Ff * Dd * 2);
  unsigned short* w2_bf    = (unsigned short*)alloc((size_t)Dd * Ff * 2);
  unsigned short* xn_bf    = (unsigned short*)alloc((size_t)M * Dd * 2);
  unsigned short* qkv_bf   = (unsigned short*)alloc((size_t)M * 3 * Dd * 2);
  unsigned short* attn_bf  = (unsigned short*)alloc((size_t)M * Dd * 2);
  float* x1                = (float*)alloc((size_t)M * Dd * 4);
  // reuse: FFN hidden (32MB) overlays dead qkv(24MB)+attn(8MB); LN2 reuses LN1
  unsigned short* h_bf   = qkv_bf;
  unsigned short* xn2_bf = xn_bf;

  // 1) weights -> bf16
  cvt_f32_bf16<<<2048, 256, 0, stream>>>(w_in, w_in_bf, (size_t)3 * Dd * Dd);
  cvt_f32_bf16<<<2048, 256, 0, stream>>>(w_out, w_out_bf, (size_t)Dd * Dd);
  cvt_f32_bf16<<<2048, 256, 0, stream>>>(w1, w1_bf, (size_t)Ff * Dd);
  cvt_f32_bf16<<<2048, 256, 0, stream>>>(w2, w2_bf, (size_t)Dd * Ff);

  // 2) zero mean-probs output (accumulated via atomics)
  (void)hipMemsetAsync(out_w, 0, (size_t)Bb * Nn * Nn * sizeof(float), stream);

  // 3) LN1
  layernorm_bf16<<<M, 256, 0, stream>>>(x, ln1g, ln1b, xn_bf, Dd);

  // 4) QKV projection: [4096,1024] x [3072,1024]^T -> bf16 [4096,3072]
  {
    dim3 grid(3 * Dd / 128, M / 64);
    gemm_bf16_kernel<<<grid, 256, 0, stream>>>(xn_bf, w_in_bf, b_in, M, 3 * Dd,
                                               Dd, 0, qkv_bf, nullptr, nullptr,
                                               nullptr);
  }

  // 5) attention (+ mean probs)
  {
    dim3 grid(Nn / 64, Hh, Bb);
    attention_kernel<<<grid, 256, 0, stream>>>(qkv_bf, kpm, attn_bf, out_w, Nn,
                                               Dd, Hh);
  }

  // 6) out-proj + pad-zero + residual -> x1 (fp32)
  {
    dim3 grid(Dd / 128, M / 64);
    gemm_bf16_kernel<<<grid, 256, 0, stream>>>(attn_bf, w_out_bf, b_out, M, Dd,
                                               Dd, 2, nullptr, x1, x, kpm);
  }

  // 7) LN2
  layernorm_bf16<<<M, 256, 0, stream>>>(x1, ln2g, ln2b, xn2_bf, Dd);

  // 8) FFN up + quickGELU -> bf16 hidden
  {
    dim3 grid(Ff / 128, M / 64);
    gemm_bf16_kernel<<<grid, 256, 0, stream>>>(xn2_bf, w1_bf, b1, M, Ff, Dd, 1,
                                               h_bf, nullptr, nullptr, nullptr);
  }

  // 9) FFN down + residual -> final output
  {
    dim3 grid(Dd / 128, M / 64);
    gemm_bf16_kernel<<<grid, 256, 0, stream>>>(h_bf, w2_bf, b2, M, Dd, Ff, 3,
                                               nullptr, out_x, x1, nullptr);
  }
}
